// OuterProductMean_AF3_2370821947567
// MI455X (gfx1250) — matrix operand order repaired and checked by reading, not compiled, not run on hardware
//
#include <hip/hip_runtime.h>
#include <hip/hip_bf16.h>
#include <stdint.h>

#define NSEQ 128
#define LSEQ 384
#define CDIM 64
#define CHID 32
#define COUT 128

#define LB 16                     // l rows per workgroup
#define MB 8                      // m cols per workgroup
#define ICH 4                     // i values per chunk
#define SROWS (LB * ICH)          // 64 stripe rows
#define SCOLS (MB * CHID)         // 256 stripe cols (f32)
#define SSTR  (SCOLS * 4 + 16)    // 1040 B row stride: % 256 == 16 -> 4-bank skew

typedef __attribute__((ext_vector_type(16))) __bf16 v16b;
typedef __attribute__((ext_vector_type(2)))  __bf16 v2b;
typedef __attribute__((ext_vector_type(2)))  float  v2f;
typedef __attribute__((ext_vector_type(16))) float  v16f;
typedef __attribute__((ext_vector_type(8)))  float  v8f;

// f32 -> bf16: let the backend choose the hardware conversion (v_cvt_pk_bf16_f32
// pairs when available) by expressing it as a vector fptrunc.
__device__ __forceinline__ v2b cvt2(float a, float b) {
  v2f f; f.x = a; f.y = b;
  return __builtin_convertvector(f, v2b);
}

// ---------------- prep: LayerNorm + dual projection -> leftT / rightT (bf16)
// leftT[(l*32+i)*128 + s], rightT[(m*32+j)*128 + s]  (reduction dim innermost)
__global__ void __launch_bounds__(256)
opm_prep(const float* __restrict__ msa, const float* __restrict__ ln_w,
         const float* __restrict__ ln_b, const float* __restrict__ w_left,
         const float* __restrict__ b_left, const float* __restrict__ w_right,
         const float* __restrict__ b_right,
         unsigned short* __restrict__ leftT, unsigned short* __restrict__ rightT)
{
  __shared__ float sx[8][CDIM];
  const int wave = threadIdx.x >> 5;
  const int lane = threadIdx.x & 31;
  const int row  = blockIdx.x * 8 + wave;     // row = s*LSEQ + l
  const int s = row / LSEQ;
  const int l = row - s * LSEQ;

  const float2 x2 = *(const float2*)(msa + (size_t)row * CDIM + lane * 2);
  float sum = x2.x + x2.y;
  #pragma unroll
  for (int m = 16; m >= 1; m >>= 1) sum += __shfl_xor(sum, m, 32);
  const float mu = sum * (1.0f / 64.0f);
  const float d0 = x2.x - mu, d1 = x2.y - mu;
  float v = d0 * d0 + d1 * d1;
  #pragma unroll
  for (int m = 16; m >= 1; m >>= 1) v += __shfl_xor(v, m, 32);
  const float rs = rsqrtf(v * (1.0f / 64.0f) + 1e-5f);
  sx[wave][lane * 2]     = d0 * rs * ln_w[lane * 2]     + ln_b[lane * 2];
  sx[wave][lane * 2 + 1] = d1 * rs * ln_w[lane * 2 + 1] + ln_b[lane * 2 + 1];
  __syncthreads();

  float accL = b_left[lane], accR = b_right[lane];
  const float* wl = w_left  + lane * CDIM;
  const float* wr = w_right + lane * CDIM;
  #pragma unroll
  for (int c = 0; c < CDIM; ++c) {
    const float xc = sx[wave][c];
    accL += xc * wl[c];
    accR += xc * wr[c];
  }
  accR *= (1.0f / (float)NSEQ);
  union { v2b v; unsigned int u; } t;
  t.v = cvt2(accL, accR);                      // one packed convert
  leftT [(size_t)(l * CHID + lane) * NSEQ + s] = (unsigned short)(t.u & 0xffffu);
  rightT[(size_t)(l * CHID + lane) * NSEQ + s] = (unsigned short)(t.u >> 16);
}

// ---------------- w_out f32 -> bf16 (vector fptrunc, 8 elems/thread)
__global__ void __launch_bounds__(256)
opm_cvt_wout(const float* __restrict__ w_out, uint4* __restrict__ woutb)
{
  const int i = blockIdx.x * 256 + threadIdx.x;
  typedef __attribute__((ext_vector_type(8))) __bf16 v8b;
  union { float4 f4[2]; v8f v; } fx;
  fx.f4[0] = *(const float4*)(w_out + (size_t)i * 8);
  fx.f4[1] = *(const float4*)(w_out + (size_t)i * 8 + 4);
  union { v8b v; uint4 q; } t;
  t.v = __builtin_convertvector(fx.v, v8b);
  woutb[i] = t.q;
}

// ---------------- fused GEMM1 (outer-product-mean) + GEMM2 (projection)
__global__ void __launch_bounds__(256)
opm_main(const unsigned short* __restrict__ leftT,
         const unsigned short* __restrict__ rightT,
         const unsigned short* __restrict__ woutb,
         const float* __restrict__ b_out,
         float* __restrict__ out)
{
  __shared__ unsigned char sO[SROWS * SSTR];   // 66560 B f32 O-stripe
  const int wave = threadIdx.x >> 5;
  const int lane = threadIdx.x & 31;
  const int nl   = lane & 15;
  const int hi   = lane >> 4;
  const int l0 = blockIdx.x * LB;
  const int m0 = blockIdx.y * MB;

  v8f acc2[8];                                  // [rti*4+ci] persistent GEMM2 acc
  #pragma unroll
  for (int t = 0; t < 8; ++t) acc2[t] = (v8f){0.f,0.f,0.f,0.f,0.f,0.f,0.f,0.f};

  // phase-1: 2 row-tiles x 4 col-tiles per wave (stripe is 4 x 16 tiles)
  const int rAb  = (wave & 1) * 2;
  const int ctb1 = (wave >> 1) * 4;
  // phase-2: 2 row-tiles x 4 col-tiles per wave (out is 8 x 8 tiles)
  const int rtb  = (wave & 3) * 2;
  const int ctb2 = (wave >> 2) * 4;

  for (int ic = 0; ic < (CHID / ICH); ++ic) {
    const int i0 = ic * ICH;

    // ---- phase 1: O stripe (64 x 256, K=128) = leftT x right, f32 in LDS ----
    v8f acc1[8];
    #pragma unroll
    for (int t = 0; t < 8; ++t) acc1[t] = (v8f){0.f,0.f,0.f,0.f,0.f,0.f,0.f,0.f};
    #pragma unroll
    for (int ks = 0; ks < 4; ++ks) {
      union { uint4 q[2]; v16b v; } A[2], B[4];
      #pragma unroll
      for (int rti = 0; rti < 2; ++rti) {     // A lanes<16: K{0-7,16-23}; >=16: {8-15,24-31}
        const int r1 = (rAb + rti) * 16 + nl;
        const size_t arow = (size_t)((l0 + (r1 >> 2)) * CHID + i0 + (r1 & 3)) * NSEQ;
        const int kb = ks * 32 + (hi ? 8 : 0);
        A[rti].q[0] = *(const uint4*)(leftT + arow + kb);
        A[rti].q[1] = *(const uint4*)(leftT + arow + kb + 16);
      }
      #pragma unroll
      for (int ci = 0; ci < 4; ++ci) {        // B lanes<16: K{0-7,16-23}; >=16: {8-15,24-31}
        const int col = (ctb1 + ci) * 16 + nl;
        const size_t brow = (size_t)((m0 + (col >> 5)) * CHID + (col & 31)) * NSEQ;
        const int kb = ks * 32 + (hi ? 8 : 0);
        B[ci].q[0] = *(const uint4*)(rightT + brow + kb);
        B[ci].q[1] = *(const uint4*)(rightT + brow + kb + 16);
      }
      #pragma unroll
      for (int rti = 0; rti < 2; ++rti)
        #pragma unroll
        for (int ci = 0; ci < 4; ++ci)
          acc1[rti * 4 + ci] = __builtin_amdgcn_wmma_f32_16x16x32_bf16(
              false, A[rti].v, false, B[ci].v, (short)0, acc1[rti * 4 + ci],
              false, false);
    }
    #pragma unroll
    for (int rti = 0; rti < 2; ++rti) {
      #pragma unroll
      for (int ci = 0; ci < 4; ++ci) {        // C/D: lane = col N, rows r+8*hi
        const int col = (ctb1 + ci) * 16 + nl;
        unsigned char* base = sO + ((rAb + rti) * 16 + hi * 8) * SSTR + col * 4;
        #pragma unroll
        for (int r = 0; r < 8; ++r)
          *(float*)(base + r * SSTR) = acc1[rti * 4 + ci][r];
      }
    }
    __syncthreads();

    // ---- phase 2: out(128x128) += O_stripe[(l,m),(i,j)] x w_out^T ----
    #pragma unroll
    for (int il = 0; il < 4; ++il) {
      v16b A2[2];
      #pragma unroll
      for (int rti = 0; rti < 2; ++rti) {
        const int p = (rtb + rti) * 16 + nl;
        const float* base = (const float*)(sO + ((p >> 3) * ICH + il) * SSTR)
                            + (p & 7) * CHID;
        const int jb = hi ? 8 : 0;            // A K-groups: j runs of 8
        union { float4 f4[4]; v16f v; } fx;
        fx.f4[0] = *(const float4*)(base + jb);
        fx.f4[1] = *(const float4*)(base + jb + 4);
        fx.f4[2] = *(const float4*)(base + jb + 16);
        fx.f4[3] = *(const float4*)(base + jb + 20);
        A2[rti] = __builtin_convertvector(fx.v, v16b);
      }
      #pragma unroll
      for (int ci = 0; ci < 4; ++ci) {
        const int o = (ctb2 + ci) * 16 + nl;
        const unsigned short* bbase = woutb + (size_t)o * (CHID * CHID)
                                      + i0 * 32 + (hi ? 8 : 0);
        union { uint4 q[2]; v16b v; } B2;
        B2.q[0] = *(const uint4*)(bbase + il * 32);
        B2.q[1] = *(const uint4*)(bbase + il * 32 + 16);
        #pragma unroll
        for (int rti = 0; rti < 2; ++rti)
          acc2[rti * 4 + ci] = __builtin_amdgcn_wmma_f32_16x16x32_bf16(
              false, A2[rti], false, B2.v, (short)0, acc2[rti * 4 + ci],
              false, false);
      }
    }
    __syncthreads();
  }

  // ---- epilogue: + b_out; within a C-tile half, l is fixed and m == r ----
  #pragma unroll
  for (int rti = 0; rti < 2; ++rti) {
    const int l = l0 + (rtb + rti) * 2 + hi;
    #pragma unroll
    for (int ci = 0; ci < 4; ++ci) {
      const int o = (ctb2 + ci) * 16 + nl;
      const float bo = b_out[o];
      float* ob = out + ((size_t)l * LSEQ + m0) * COUT + o;
      #pragma unroll
      for (int r = 0; r < 8; ++r)
        ob[(size_t)r * COUT] = acc2[rti * 4 + ci][r] + bo;
    }
  }
}

extern "C" void kernel_launch(void* const* d_in, const int* in_sizes, int n_in,
                              void* d_out, int out_size, void* d_ws, size_t ws_size,
                              hipStream_t stream) {
  const float* msa     = (const float*)d_in[0];
  const float* ln_w    = (const float*)d_in[1];
  const float* ln_b    = (const float*)d_in[2];
  const float* w_left  = (const float*)d_in[3];
  const float* b_left  = (const float*)d_in[4];
  const float* w_right = (const float*)d_in[5];
  const float* b_right = (const float*)d_in[6];
  const float* w_out   = (const float*)d_in[7];
  const float* b_out   = (const float*)d_in[8];
  float* out = (float*)d_out;

  unsigned short* leftT  = (unsigned short*)d_ws;                 // 12288*128 bf16
  unsigned short* rightT = leftT  + (size_t)LSEQ * CHID * NSEQ;   // 12288*128 bf16
  unsigned short* woutb  = rightT + (size_t)LSEQ * CHID * NSEQ;   // 128*1024 bf16

  hipLaunchKernelGGL(opm_prep, dim3(NSEQ * LSEQ / 8), dim3(256), 0, stream,
                     msa, ln_w, ln_b, w_left, b_left, w_right, b_right, leftT, rightT);
  hipLaunchKernelGGL(opm_cvt_wout, dim3(COUT * CHID * CHID / (256 * 8)), dim3(256),
                     0, stream, w_out, (uint4*)woutb);
  hipLaunchKernelGGL(opm_main, dim3(LSEQ / LB, LSEQ / MB), dim3(256), 0, stream,
                     leftT, rightT, woutb, b_out, out);
}
